// VideoPrismSelfAttention_86457691669095
// MI455X (gfx1250) — matrix operand order repaired and checked by reading, not compiled, not run on hardware
//
#include <hip/hip_runtime.h>
#include <hip/hip_bf16.h>
#include <math.h>

typedef __attribute__((ext_vector_type(16))) __bf16 v16bf;
typedef __attribute__((ext_vector_type(8)))  float  v8f;
typedef __attribute__((ext_vector_type(2)))  __bf16 bf16x2;

#define SOFTCAP     50.0f
#define INV_SOFTCAP (1.0f / 50.0f)
#define SCALING     0.125f   // 64^-0.5

static __device__ __forceinline__ v8f wmma_bf16(v16bf a, v16bf b, v8f c) {
    // D = A(16x32 bf16) * B(32x16 bf16) + C(16x16 f32)
    return __builtin_amdgcn_wmma_f32_16x16x32_bf16(false, a, false, b, (short)0, c, false, false);
}

// ---- Fragment loaders (layouts per CDNA5 ISA §7.12.2) ----
// A fragment 16(M)x32(K); src points at tile (0,0); element (m,k) at src[m*ld + k]
static __device__ __forceinline__ v16bf load_a_bf16(const __bf16* src, int ld) {
    const int lane = threadIdx.x & 31;
    const int row = lane & 15, hf = lane >> 4;
    const __bf16* p = src + row * ld;
    v16bf a;
#pragma unroll
    for (int v = 0; v < 8; ++v) {
        const int kpos = ((v >> 2) << 4) + (hf << 3) + ((v & 3) << 1);
        const bf16x2 f = *(const bf16x2*)(p + kpos);
        a[2 * v]     = f[0];
        a[2 * v + 1] = f[1];
    }
    return a;
}

// B fragment 32(K)x16(N); element (k,n) stored at src[n*ld + k] (N slow, K fast)
static __device__ __forceinline__ v16bf load_b_bf16(const __bf16* src, int ld) {
    const int lane = threadIdx.x & 31;
    const int n = lane & 15, hf = lane >> 4;
    const __bf16* p = src + n * ld;
    v16bf b;
#pragma unroll
    for (int v = 0; v < 8; ++v) {
        const int kpos = ((v >> 2) << 4) + (hf << 3) + ((v & 3) << 1);
        const bf16x2 f = *(const bf16x2*)(p + kpos);
        b[2 * v]     = f[0];
        b[2 * v + 1] = f[1];
    }
    return b;
}

// ---- GEMM helpers: one k-step fragment set (2 A-frags + 4 B-frags) ----
static __device__ __forceinline__ void load_set(const __bf16* __restrict__ Arow,
                                                const __bf16* __restrict__ Wrow, int k0,
                                                v16bf& a0, v16bf& a1, v16bf* bfr) {
    a0 = load_a_bf16(Arow + k0, 768);
    a1 = load_a_bf16(Arow + (size_t)16 * 768 + k0, 768);
#pragma unroll
    for (int nt = 0; nt < 4; ++nt)
        bfr[nt] = load_b_bf16(Wrow + (size_t)nt * 16 * 768 + k0, 768);
}

static __device__ __forceinline__ void compute_set(const v16bf& a0, const v16bf& a1,
                                                   const v16bf* bfr, v8f acc[2][4]) {
#pragma unroll
    for (int nt = 0; nt < 4; ++nt) {
        acc[0][nt] = wmma_bf16(a0, bfr[nt], acc[0][nt]);
        acc[1][nt] = wmma_bf16(a1, bfr[nt], acc[1][nt]);
    }
}

// ================== Kernel 0: fp32 -> bf16 bulk convert ==================
__global__ __launch_bounds__(256)
void f32_to_bf16_kernel(const float* __restrict__ src, __bf16* __restrict__ dst, int n) {
    const int i = (blockIdx.x * 256 + threadIdx.x) * 8;
    if (i + 8 <= n) {
        const float4 a = *(const float4*)(src + i);
        const float4 b = *(const float4*)(src + i + 4);
        bf16x2 o[4];
        o[0][0] = (__bf16)a.x; o[0][1] = (__bf16)a.y;
        o[1][0] = (__bf16)a.z; o[1][1] = (__bf16)a.w;
        o[2][0] = (__bf16)b.x; o[2][1] = (__bf16)b.y;
        o[3][0] = (__bf16)b.z; o[3][1] = (__bf16)b.w;
#pragma unroll
        for (int j = 0; j < 4; ++j) *(bf16x2*)(dst + i + 2 * j) = o[j];
    } else {
        for (int j = i; j < n; ++j) dst[j] = (__bf16)src[j];
    }
}

// ================== Kernel A: fused Q/K/V projection ==================
// 32(M)x64(N) tile per wave; k-loop unrolled x2 with ping-pong buffer sets
// (no register-rotation movs). grid: 2304 blocks * 4 waves = 9216 waves.
__global__ __launch_bounds__(128)
void qkv_proj_kernel(const __bf16* __restrict__ Xbf,
                     const __bf16* __restrict__ wqb, const float* __restrict__ bq,
                     const __bf16* __restrict__ wkb, const float* __restrict__ bk,
                     const __bf16* __restrict__ wvb, const float* __restrict__ bvv,
                     __bf16* __restrict__ Qbf, __bf16* __restrict__ Kbf,
                     __bf16* __restrict__ Vt) {
    const int wave = threadIdx.x >> 5;
    const int gid  = blockIdx.x * 4 + wave;
    const int qkv  = gid / (256 * 12);
    const int rem  = gid % (256 * 12);
    const int mt   = rem / 12;
    const int ns   = rem % 12;
    const int lane = threadIdx.x & 31;
    const int lanelo = lane & 15, hf = lane >> 4;

    const __bf16* W   = (qkv == 0) ? wqb : (qkv == 1) ? wkb : wvb;
    const float* bias = (qkv == 0) ? bq : (qkv == 1) ? bk : bvv;

    const v8f zero = {0, 0, 0, 0, 0, 0, 0, 0};
    v8f acc[2][4];
#pragma unroll
    for (int mi = 0; mi < 2; ++mi)
#pragma unroll
        for (int nt = 0; nt < 4; ++nt) acc[mi][nt] = zero;

    const __bf16* Xrow = Xbf + (size_t)mt * 32 * 768;
    const __bf16* Wrow = W + (size_t)ns * 64 * 768;

    // 24 k-steps of 32; ping-pong sets A/B, 2 steps per loop iteration
    v16bf a0A, a1A, bfA[4], a0B, a1B, bfB[4];
    load_set(Xrow, Wrow, 0, a0A, a1A, bfA);
    for (int ks = 32; ks < 704; ks += 64) {
        load_set(Xrow, Wrow, ks, a0B, a1B, bfB);
        compute_set(a0A, a1A, bfA, acc);
        load_set(Xrow, Wrow, ks + 32, a0A, a1A, bfA);
        compute_set(a0B, a1B, bfB, acc);
    }
    load_set(Xrow, Wrow, 736, a0B, a1B, bfB);
    compute_set(a0A, a1A, bfA, acc);   // k-step 704
    compute_set(a0B, a1B, bfB, acc);   // k-step 736

#pragma unroll
    for (int nt = 0; nt < 4; ++nt) {
        const int e  = ns * 64 + nt * 16 + lanelo;
        const float bval = bias[e];
        const int h  = e >> 6, hd = e & 63;
#pragma unroll
        for (int mi = 0; mi < 2; ++mi) {
#pragma unroll
            for (int r = 0; r < 8; ++r) {
                const int m = mt * 32 + mi * 16 + r + (hf << 3);  // global row in [B*S]
                const int bb = m >> 11, s = m & 2047;
                const float val = acc[mi][nt][r] + bval;
                if (qkv == 0) {
                    Qbf[(((size_t)(bb * 12 + h) * 2048) + s) * 64 + hd] = (__bf16)(val * SCALING);
                } else if (qkv == 1) {
                    Kbf[(((size_t)(bb * 12 + h) * 2048) + s) * 64 + hd] = (__bf16)val;
                } else {
                    Vt[((size_t)(bb * 12 + h) * 64 + hd) * 2048 + s] = (__bf16)val;  // V^T
                }
            }
        }
    }
}

// ================== Kernel B: flash attention w/ tanh softcap ==================
// grid: 1536 blocks * 4 waves = 6144 waves = 48 (b,h) * 128 (16-row q tiles).
// 128 q-tiles per bh and 4 waves/block => all 4 waves of a block share one (b,h),
// so K/V tiles are cooperatively staged in LDS once per block.
__global__ __launch_bounds__(128)
void attn_kernel(const float* __restrict__ mask,
                 const __bf16* __restrict__ Qbf, const __bf16* __restrict__ Kbf,
                 const __bf16* __restrict__ Vt, __bf16* __restrict__ Ctx) {
    __shared__ __align__(16) __bf16 ldsK[32 * 64];      // [key][hd]
    __shared__ __align__(16) __bf16 ldsV[64 * 32];      // [hd][key]
    __shared__ __align__(16) __bf16 pLds[4][16 * 32];   // per-wave P relayout
    const int tid  = threadIdx.x;
    const int wave = tid >> 5;
    const int gid  = blockIdx.x * 4 + wave;
    const int bh   = gid >> 7;
    const int qt   = gid & 127;
    const int b    = bh / 12, h = bh % 12;
    const int lane = tid & 31, lanelo = lane & 15, hf = lane >> 4;
    const int sq   = qt * 16;

    const __bf16* Qp = Qbf + ((size_t)bh * 2048 + sq) * 64;
    const v16bf aq_lo = load_a_bf16(Qp, 64);        // head-dim 0..31
    const v16bf aq_hi = load_a_bf16(Qp + 32, 64);   // head-dim 32..63

    const v8f zero = {0, 0, 0, 0, 0, 0, 0, 0};
    v8f acc[4];
#pragma unroll
    for (int i = 0; i < 4; ++i) acc[i] = zero;
    float m_i[8], l_i[8];
#pragma unroll
    for (int r = 0; r < 8; ++r) { m_i[r] = -INFINITY; l_i[r] = 0.0f; }

    const float* maskRow = mask + ((size_t)b * 2048 + sq) * 2048;
    __bf16* myLds = &pLds[wave][0];
    // block-uniform source bases for staging
    const uint4* Ksrc4 = (const uint4*)(Kbf + (size_t)bh * 2048 * 64);   // 8 uint4 per key row

    for (int kb = 0; kb < 64; ++kb) {
        const int s0 = kb * 32;
        // ---- cooperative staging: K tile (contiguous 4KB) + V^T tile (64 rows x 64B)
#pragma unroll
        for (int c = 0; c < 2; ++c) {
            const int idx = tid + c * 128;                       // 0..255
            ((uint4*)ldsK)[idx] = Ksrc4[(size_t)s0 * 8 + idx];
            const int hd = idx >> 2, j = idx & 3;
            ((uint4*)ldsV)[idx] =
                *(const uint4*)(Vt + ((size_t)bh * 64 + hd) * 2048 + s0 + j * 8);
        }
        __syncthreads();

        // ---- scores: 16 q-rows x 32 keys (2 tiles, K-dim=64 via 2 wmma each)
        v8f sc[2];
#pragma unroll
        for (int nt = 0; nt < 2; ++nt) {
            const __bf16* Kp = ldsK + (nt * 16) * 64;
            v8f c = wmma_bf16(aq_lo, load_b_bf16(Kp, 64), zero);
            c = wmma_bf16(aq_hi, load_b_bf16(Kp + 32, 64), c);
            sc[nt] = c;
        }
        // ---- softcap + mask + online softmax (fp32)
#pragma unroll
        for (int r = 0; r < 8; ++r) {
            const int qrow = r + (hf << 3);
            float p0 = sc[0][r], p1 = sc[1][r];
            p0 = tanhf(p0 * INV_SOFTCAP) * SOFTCAP;
            p1 = tanhf(p1 * INV_SOFTCAP) * SOFTCAP;
            p0 += maskRow[(size_t)qrow * 2048 + (s0 + lanelo)];
            p1 += maskRow[(size_t)qrow * 2048 + (s0 + 16 + lanelo)];
            float mx = fmaxf(p0, p1);
            mx = fmaxf(mx, __shfl_xor(mx, 1, 32));
            mx = fmaxf(mx, __shfl_xor(mx, 2, 32));
            mx = fmaxf(mx, __shfl_xor(mx, 4, 32));
            mx = fmaxf(mx, __shfl_xor(mx, 8, 32));
            const float m_new = fmaxf(m_i[r], mx);
            const float alpha = __expf(m_i[r] - m_new);
            m_i[r] = m_new;
            p0 = __expf(p0 - m_new);
            p1 = __expf(p1 - m_new);
            float sum = p0 + p1;
            sum += __shfl_xor(sum, 1, 32);
            sum += __shfl_xor(sum, 2, 32);
            sum += __shfl_xor(sum, 4, 32);
            sum += __shfl_xor(sum, 8, 32);
            l_i[r] = l_i[r] * alpha + sum;
#pragma unroll
            for (int nt = 0; nt < 4; ++nt) acc[nt][r] *= alpha;
            // C-layout -> A-fragment layout via per-wave LDS round trip
            myLds[qrow * 32 + lanelo]      = (__bf16)p0;
            myLds[qrow * 32 + 16 + lanelo] = (__bf16)p1;
        }
        asm volatile("s_wait_dscnt 0x0" ::: "memory");
        const v16bf pa = load_a_bf16(myLds, 32);   // P: 16x32 bf16
        // ---- PV: P(16x32) * V(32x64) from LDS [hd][key]
#pragma unroll
        for (int nt = 0; nt < 4; ++nt) {
            acc[nt] = wmma_bf16(pa, load_b_bf16(ldsV + (nt * 16) * 32, 32), acc[nt]);
        }
        __syncthreads();   // protect ldsK/ldsV before next iteration's staging
    }
    // ---- finalize + store ctx [B*S][768] bf16, col = h*64 + hd
#pragma unroll
    for (int r = 0; r < 8; ++r) {
        const int grow = sq + r + (hf << 3);
        const float inv = 1.0f / l_i[r];
#pragma unroll
        for (int nt = 0; nt < 4; ++nt) {
            const int col = h * 64 + nt * 16 + lanelo;
            Ctx[((size_t)b * 2048 + grow) * 768 + col] = (__bf16)(acc[nt][r] * inv);
        }
    }
}

// ================== Kernel C: output projection ==================
// 32(M)x64(N) tile per wave; k-loop unrolled x2 with ping-pong buffer sets.
// grid: 768 blocks * 4 waves = 3072 waves = 256 (M tiles) * 12 (N strips)
__global__ __launch_bounds__(128)
void out_proj_kernel(const __bf16* __restrict__ Ctx, const __bf16* __restrict__ wob,
                     const float* __restrict__ bo, float* __restrict__ Out) {
    const int wave = threadIdx.x >> 5;
    const int gid  = blockIdx.x * 4 + wave;
    const int mt   = gid / 12;
    const int ns   = gid % 12;
    const int lane = threadIdx.x & 31, lanelo = lane & 15, hf = lane >> 4;

    const v8f zero = {0, 0, 0, 0, 0, 0, 0, 0};
    v8f acc[2][4];
#pragma unroll
    for (int mi = 0; mi < 2; ++mi)
#pragma unroll
        for (int nt = 0; nt < 4; ++nt) acc[mi][nt] = zero;

    const __bf16* Crow = Ctx + (size_t)mt * 32 * 768;
    const __bf16* Wrow = wob + (size_t)ns * 64 * 768;

    v16bf a0A, a1A, bfA[4], a0B, a1B, bfB[4];
    load_set(Crow, Wrow, 0, a0A, a1A, bfA);
    for (int ks = 32; ks < 704; ks += 64) {
        load_set(Crow, Wrow, ks, a0B, a1B, bfB);
        compute_set(a0A, a1A, bfA, acc);
        load_set(Crow, Wrow, ks + 32, a0A, a1A, bfA);
        compute_set(a0B, a1B, bfB, acc);
    }
    load_set(Crow, Wrow, 736, a0B, a1B, bfB);
    compute_set(a0A, a1A, bfA, acc);
    compute_set(a0B, a1B, bfB, acc);

#pragma unroll
    for (int nt = 0; nt < 4; ++nt) {
        const int e = ns * 64 + nt * 16 + lanelo;
        const float bval = bo[e];
#pragma unroll
        for (int mi = 0; mi < 2; ++mi) {
#pragma unroll
            for (int r = 0; r < 8; ++r) {
                const int m = mt * 32 + mi * 16 + r + (hf << 3);
                Out[(size_t)m * 768 + e] = acc[mi][nt][r] + bval;
            }
        }
    }
}

extern "C" void kernel_launch(void* const* d_in, const int* in_sizes, int n_in,
                              void* d_out, int out_size, void* d_ws, size_t ws_size,
                              hipStream_t stream) {
    const float* X    = (const float*)d_in[0];
    const float* mask = (const float*)d_in[1];
    const float* wq   = (const float*)d_in[2];
    const float* bq   = (const float*)d_in[3];
    const float* wk   = (const float*)d_in[4];
    const float* bk   = (const float*)d_in[5];
    const float* wv   = (const float*)d_in[6];
    const float* bv   = (const float*)d_in[7];
    const float* wo   = (const float*)d_in[8];
    const float* bo   = (const float*)d_in[9];
    float* out = (float*)d_out;

    const size_t nX = (size_t)8192 * 768;   // B*S*D = 6291456
    const size_t nW = (size_t)768 * 768;    // 589824
    __bf16* Xbf  = (__bf16*)d_ws;
    __bf16* Wqb  = Xbf + nX;
    __bf16* Wkb  = Wqb + nW;
    __bf16* Wvb  = Wkb + nW;
    __bf16* Wob  = Wvb + nW;
    __bf16* Qbf  = Wob + nW;                // [B,H,S,HD] pre-scaled
    __bf16* Kbf  = Qbf + nX;                // [B,H,S,HD]
    __bf16* Vt   = Kbf + nX;                // [B,H,HD,S] (transposed)
    __bf16* Ctx  = Vt + nX;                 // [B*S, D]

    // one-time bf16 conversion (X + 4 weights)
    f32_to_bf16_kernel<<<(int)(nX / (256 * 8)), 256, 0, stream>>>(X, Xbf, (int)nX);
    f32_to_bf16_kernel<<<(int)(nW / (256 * 8)), 256, 0, stream>>>(wq, Wqb, (int)nW);
    f32_to_bf16_kernel<<<(int)(nW / (256 * 8)), 256, 0, stream>>>(wk, Wkb, (int)nW);
    f32_to_bf16_kernel<<<(int)(nW / (256 * 8)), 256, 0, stream>>>(wv, Wvb, (int)nW);
    f32_to_bf16_kernel<<<(int)(nW / (256 * 8)), 256, 0, stream>>>(wo, Wob, (int)nW);

    qkv_proj_kernel<<<2304, 128, 0, stream>>>(Xbf, Wqb, bq, Wkb, bk, Wvb, bv, Qbf, Kbf, Vt);
    attn_kernel<<<1536, 128, 0, stream>>>(mask, Qbf, Kbf, Vt, Ctx);
    out_proj_kernel<<<768, 128, 0, stream>>>(Ctx, Wob, bo, out);
}
